// Prop_0129_86870008529148
// MI455X (gfx1250) — matrix operand order repaired and checked by reading, not compiled, not run on hardware
//
#include <hip/hip_runtime.h>

// ---------------------------------------------------------------------------
// Conv1d autoencoder on MI455X (gfx1250): every conv is an implicit GEMM on
// V_WMMA_F32_16X16X32_BF16.  bf16 activations (each layer's 42MB tensor lives
// in the 192MB L2 between layers), f32 accumulation, fused
// bias/LeakyReLU/tanh/residual epilogue.
//
// K ordering is tap-major: kappa = r*Cpad + ci (Cpad = Cin rounded to even).
// Raw input tile is staged TRANSPOSED in LDS (raw[t][ci]), so each im2col
// B-fragment dword (two adjacent channels at one tap) is a single aligned
// 32-bit LDS load.  Weights are pre-packed on device into the ISA A-fragment
// lane layout so each lane's A fragment is one 32B global load.
// Tile/epilogue control flow is forced wave-uniform via readfirstlane so the
// full-M-tile store path is branchless scalar-steered code.
// ---------------------------------------------------------------------------

#define BB 16
#define TT 32768
#define TB (BB * TT) // 524288

typedef __attribute__((ext_vector_type(16))) __bf16 v16bf;
typedef __attribute__((ext_vector_type(8))) float v8f;

__device__ __forceinline__ float bf2f(unsigned short u) {
  unsigned int v = ((unsigned int)u) << 16;
  float f;
  __builtin_memcpy(&f, &v, 4);
  return f;
}
__device__ __forceinline__ unsigned short f2bf(float f) {
  unsigned int v;
  __builtin_memcpy(&v, &f, 4);
  unsigned int lsb = (v >> 16) & 1u;
  v += 0x7fffu + lsb; // round-to-nearest-even
  return (unsigned short)(v >> 16);
}

// ---------------------------------------------------------------------------
// Pack conv weights (f32 [Cout][Cin][9]) into WMMA A-fragment lane order with
// kappa = r*Cpad + ci.  16-bit A 16x32 layout: lanes 0-15 = rows M,
// VGPR j -> K = 16*kh + 2j (contiguous, matches B fragment), kh=lane>>4.
// One dword = K pair; chunk = (mtile, kchunk) = 32 lanes * 8 dwords.
// ---------------------------------------------------------------------------
__global__ void prepack_w(const float* __restrict__ w, unsigned int* __restrict__ ap,
                          int Cin, int Cpad, int Cout, int KC) {
  int e = blockIdx.x * 256 + threadIdx.x; // grid covers MT*KC*256 exactly
  int chunk = e >> 8;
  int rr = e & 255;
  int lane = rr >> 3, j = rr & 7;
  int mt = chunk / KC, kc = chunk - mt * KC;
  int m = mt * 16 + (lane & 15);
  int kh = lane >> 4;
  int koff = kh * 16 + 2 * j;
  int k0 = kc * 32 + koff;
  unsigned int out = 0;
#pragma unroll
  for (int s = 0; s < 2; ++s) {
    int k = k0 + s;
    unsigned short hv = 0;
    int r = k / Cpad, ci = k - r * Cpad;
    if (m < Cout && r < 9 && ci < Cin) hv = f2bf(w[(m * Cin + ci) * 9 + r]);
    out |= ((unsigned int)hv) << (16 * s);
  }
  ap[(long)chunk * 256 + lane * 8 + j] = out;
}

__global__ void f32_to_bf16(const float* __restrict__ in, unsigned short* __restrict__ out, int n) {
  int i = blockIdx.x * 256 + threadIdx.x;
  if (i < n) out[i] = f2bf(in[i]);
}

// ---------------------------------------------------------------------------
// Fused conv1d (k=9, dilation DIL, 'same' zero padding) as implicit GEMM.
// Block = 256 threads (8 wave32), tile = Cout x 64 time samples, batch b.
// ---------------------------------------------------------------------------
template <int CIN, int COUT, int DIL, int ACT, bool RES, bool OUT32>
__global__ __launch_bounds__(256) void conv1d_wmma(
    const unsigned short* __restrict__ in, long in_bs,
    const unsigned int* __restrict__ ap, const float* __restrict__ bias,
    void* __restrict__ outp, long out_bs,
    const unsigned short* __restrict__ res, long res_bs) {
  constexpr int CP = (CIN + 1) & ~1;       // channels padded to even
  constexpr int KK = 9 * CP;               // logical K extent
  constexpr int KC = (KK + 31) / 32;       // K chunks of 32
  constexpr int KP2 = KC * 16;             // padded K pairs (dwords/column)
  constexpr int CSTR = KP2 + 8;            // column stride dwords (32B aligned)
  constexpr int RS = 64 + 8 * DIL;         // raw tile width incl halo (even)
  constexpr int CS = CP + 2;               // raw row stride (u16, even, de-banked)
  constexpr int HALO = 4 * DIL;
  constexpr int MT = (COUT + 15) / 16;
  constexpr bool HASFULL = (COUT >= 16);   // any full 16-row M tiles exist?

  extern __shared__ __align__(32) char smem[];
  __shared__ __attribute__((aligned(32))) float sbias[MT * 16];
  unsigned short* raw = (unsigned short*)smem;                    // [RS][CS]
  constexpr int RAWB = (RS * CS * 2 + 31) & ~31;
  unsigned int* bcol = (unsigned int*)(smem + RAWB);              // [64][CSTR]

  const int tid = threadIdx.x;
  const int b = blockIdx.y;
  const int t0 = blockIdx.x * 64;

  if (tid < MT * 16) sbias[tid] = (tid < COUT) ? bias[tid] : 0.f;

  // Stage A: raw tile (transposed: [t][ci]) with zero-padded halo + channels.
  // One u32 global load per t-pair (all offsets even -> never straddles).
  for (int idx = tid; idx < CP * (RS / 2); idx += 256) {
    int ci = idx / (RS / 2), tp = idx - ci * (RS / 2);
    int t = t0 + 2 * tp - HALO;
    unsigned int v = 0;
    if (ci < CIN && t >= 0 && t < TT)
      v = *(const unsigned int*)(in + b * in_bs + (long)ci * TT + t);
    raw[(2 * tp) * CS + ci] = (unsigned short)(v & 0xffffu);
    raw[(2 * tp + 1) * CS + ci] = (unsigned short)(v >> 16);
  }
  __syncthreads();

  // Stage B: im2col into B-fragment layout [n][kpair].  Pair kp = (kappa=2kp,
  // 2kp+1) = channels (2cp,2cp+1) at tap r -> one aligned u32 LDS load.
  const unsigned int* raw32 = (const unsigned int*)raw;
  for (int idx = tid; idx < 64 * KP2; idx += 256) {
    int n = idx / KP2, kp = idx - n * KP2;
    unsigned int v = 0;
    if (kp < 9 * (CP / 2)) {
      int r = kp / (CP / 2), cp = kp - r * (CP / 2);
      v = raw32[(n + r * DIL) * (CS / 2) + cp];
    }
    bcol[n * CSTR + kp] = v;
  }
  __syncthreads();

  // Stage C: per-wave 16x16 tiles, K-loop of v_wmma_f32_16x16x32_bf16.
  // Wave index via readfirstlane -> tile/mt/nt are provably wave-uniform
  // (SGPRs), so tile-level control flow is scalar branches, not exec masks.
  const int wv = __builtin_amdgcn_readfirstlane(tid >> 5);
  const int lane = tid & 31;
  const int nl = lane & 15, kh = lane >> 4;
  for (int tile = wv; tile < MT * 4; tile += 8) {
    int mt = tile >> 2, nt = tile & 3;
    v8f acc = {};
    const unsigned int* apb = ap + (long)(mt * KC) * 256 + lane * 8;
    const unsigned int* bcb = bcol + (nt * 16 + nl) * CSTR + kh * 8;
#pragma unroll
    for (int kc = 0; kc < KC; ++kc) {
      v16bf a = *(const v16bf*)(apb + kc * 256); // 32B global, A lane frag
      v16bf bm = *(const v16bf*)(bcb + kc * 16); // 32B LDS,    B lane frag
      acc = __builtin_amdgcn_wmma_f32_16x16x32_bf16(false, a, false, bm,
                                                    (short)0, acc, false, false);
    }
    // C/D layout: VGPR v -> M = mbase + v, N = lane&15
    int n = t0 + nt * 16 + nl;
    int mbase = mt * 16 + 8 * kh;
    v8f bv = *(const v8f*)&sbias[mbase]; // one aligned 32B LDS load
    v8f ov;
#pragma unroll
    for (int v = 0; v < 8; ++v) {
      float val = acc[v] + bv[v];
      if constexpr (ACT == 1) val = (val >= 0.f) ? val : 0.01f * val;
      if constexpr (ACT == 2) val = tanhf(val);
      ov[v] = val;
    }
    const unsigned short* rp = res + (RES ? (b * res_bs + n + (long)mbase * TT) : 0);
    if (HASFULL && mt * 16 + 16 <= COUT) { // scalar branch (mt uniform)
      // Full M tile: branchless, immediate-offset addressing (v*TT*2 < 2^23).
      if constexpr (OUT32) {
        float* op = (float*)outp + b * out_bs + n + (long)mbase * TT;
#pragma unroll
        for (int v = 0; v < 8; ++v) {
          float val = ov[v];
          if constexpr (RES) val += bf2f(rp[(long)v * TT]);
          op[(long)v * TT] = val;
        }
      } else {
        unsigned short* op = (unsigned short*)outp + b * out_bs + n + (long)mbase * TT;
#pragma unroll
        for (int v = 0; v < 8; ++v) {
          float val = ov[v];
          if constexpr (RES) val += bf2f(rp[(long)v * TT]);
          op[(long)v * TT] = f2bf(val);
        }
      }
    } else {
      // Partial M tile (at most one per wave): per-element mask.
#pragma unroll
      for (int v = 0; v < 8; ++v) {
        if (mbase + v < COUT) {
          float val = ov[v];
          if constexpr (RES) val += bf2f(rp[(long)v * TT]);
          long o = b * out_bs + n + (long)(mbase + v) * TT;
          if constexpr (OUT32) ((float*)outp)[o] = val;
          else ((unsigned short*)outp)[o] = f2bf(val);
        }
      }
    }
  }
}

// ---------------------------------------------------------------------------
// Soft VQ: dist(b,t,k) = |c|^2 + |m_k|^2 - 2 c.m_k ; softmax over k=0..31;
// prob -> d_out (f32), quantized code overwrites input in place (bf16).
// ---------------------------------------------------------------------------
__global__ __launch_bounds__(256) void vq_soft(unsigned short* __restrict__ code, long code_bs,
                                               const float* __restrict__ means,
                                               float* __restrict__ prob) {
  __shared__ float msh[15 * 32];
  __shared__ float m2[32];
  int tid = threadIdx.x;
  if (tid < 480) msh[tid] = means[tid];
  __syncthreads();
  if (tid < 32) {
    float s = 0.f;
    for (int d = 0; d < 15; ++d) { float m = msh[d * 32 + tid]; s += m * m; }
    m2[tid] = s;
  }
  __syncthreads();
  long idx = (long)blockIdx.x * 256 + tid; // [0, TB)
  int b = (int)(idx / TT);
  int t = (int)(idx - (long)b * TT);
  long base = b * code_bs + t;
  float c[15], c2 = 0.f;
#pragma unroll
  for (int d = 0; d < 15; ++d) { c[d] = bf2f(code[base + (long)d * TT]); c2 += c[d] * c[d]; }
  float lg[32], mx = -1e30f;
#pragma unroll
  for (int k = 0; k < 32; ++k) {
    float dot = 0.f;
#pragma unroll
    for (int d = 0; d < 15; ++d) dot += c[d] * msh[d * 32 + k];
    lg[k] = -(c2 + m2[k] - 2.f * dot); // SCALE = 1
    mx = fmaxf(mx, lg[k]);
  }
  float s = 0.f;
#pragma unroll
  for (int k = 0; k < 32; ++k) { lg[k] = __expf(lg[k] - mx); s += lg[k]; }
  float inv = 1.f / s;
#pragma unroll
  for (int k = 0; k < 32; ++k) { lg[k] *= inv; prob[idx * 32 + k] = lg[k]; }
#pragma unroll
  for (int d = 0; d < 15; ++d) {
    float q = 0.f;
#pragma unroll
    for (int k = 0; k < 32; ++k) q += lg[k] * msh[d * 32 + k];
    code[base + (long)d * TT] = f2bf(q);
  }
}

// ---------------------------------------------------------------------------
template <int CIN, int COUT, int DIL, int ACT, bool RES, bool OUT32>
static inline void run_conv(const unsigned short* ip, long ibs,
                            const unsigned int* ap, const float* bias,
                            void* op, long obs,
                            const unsigned short* rp, long rbs, hipStream_t stream) {
  constexpr int CP = (CIN + 1) & ~1;
  constexpr int KC = (9 * CP + 31) / 32;
  constexpr int CSTR = KC * 16 + 8;
  constexpr int RS = 64 + 8 * DIL;
  constexpr int CS = CP + 2;
  constexpr size_t SH = (size_t)((RS * CS * 2 + 31) & ~31) + (size_t)64 * CSTR * 4;
  conv1d_wmma<CIN, COUT, DIL, ACT, RES, OUT32>
      <<<dim3(TT / 64, BB), dim3(256), SH, stream>>>(ip, ibs, ap, bias, op, obs, rp, rbs);
}

extern "C" void kernel_launch(void* const* d_in, const int* in_sizes, int n_in,
                              void* d_out, int out_size, void* d_ws, size_t ws_size,
                              hipStream_t stream) {
  (void)in_sizes; (void)n_in; (void)out_size; (void)ws_size;
  const float* x = (const float*)d_in[0];
  const float* mean_s = (const float*)d_in[1];
  const float* mean_n = (const float*)d_in[2];

  // d_in order: x, mean_s, mean_n, then params in setup_inputs insertion order,
  // each conv contributing (w, b).
  struct LW { int wi, bi, Cin, Cout; };
  static const LW LWT[20] = {
      {3, 4, 1, 40},    {5, 6, 40, 20},   {7, 8, 20, 40},   {9, 10, 40, 20},
      {11, 12, 20, 40}, {13, 14, 20, 10}, {15, 16, 10, 20}, {17, 18, 20, 10},
      {19, 20, 10, 20}, {21, 22, 20, 15}, {23, 24, 15, 40}, {25, 26, 40, 20},
      {27, 28, 20, 40}, {29, 30, 40, 20}, {31, 32, 20, 40}, {33, 34, 40, 20},
      {35, 36, 20, 40}, {37, 38, 40, 20}, {39, 40, 20, 40}, {41, 42, 40, 1},
  };
  const long APSTRIDE = 3L * 12 * 256; // max MT*KC chunks * 256 dwords

  char* ws = (char*)d_ws;
  unsigned int* apack = (unsigned int*)ws;
  size_t off = ((size_t)20 * APSTRIDE * 4 + 255) & ~(size_t)255;
  unsigned short* X1 = (unsigned short*)(ws + off);
  off += (size_t)TB * 2; off = (off + 255) & ~(size_t)255;
  unsigned short* Bf[4];
  for (int i = 0; i < 4; ++i) {
    Bf[i] = (unsigned short*)(ws + off);
    off += (size_t)40 * TB * 2; off = (off + 255) & ~(size_t)255;
  }
  unsigned short* CS_ = (unsigned short*)(ws + off);
  off += (size_t)15 * TB * 2; off = (off + 255) & ~(size_t)255;
  unsigned short* CN_ = (unsigned short*)(ws + off);

  float* out = (float*)d_out;
  float* s_hat = out;
  float* n_hat = out + TB;
  float* prob_s = out + 2L * TB;
  float* prob_n = out + 2L * TB + 32L * TB;

  // Pre-pack all weights into WMMA A-fragment order (tap-major kappa)
  for (int l = 0; l < 20; ++l) {
    int cp = (LWT[l].Cin + 1) & ~1;
    int MT = (LWT[l].Cout + 15) / 16, KC = (9 * cp + 31) / 32;
    prepack_w<<<dim3(MT * KC), dim3(256), 0, stream>>>(
        (const float*)d_in[LWT[l].wi], apack + l * APSTRIDE, LWT[l].Cin, cp, LWT[l].Cout, KC);
  }
  f32_to_bf16<<<dim3(TB / 256), dim3(256), 0, stream>>>(x, X1, TB);

  const long BS40 = 40L * TT, BS15 = 15L * TT, BS1 = TT;
  auto AP = [&](int l) { return apack + l * APSTRIDE; };
  auto BI = [&](int l) { return (const float*)d_in[LWT[l].bi]; };

  // encoder 1
  run_conv<1, 40, 1, 1, false, false>(X1, BS1, AP(0), BI(0), Bf[0], BS40, nullptr, 0, stream);
  run_conv<40, 20, 1, 1, false, false>(Bf[0], BS40, AP(1), BI(1), Bf[1], BS40, nullptr, 0, stream);
  run_conv<20, 40, 1, 1, true, false>(Bf[1], BS40, AP(2), BI(2), Bf[2], BS40, Bf[0], BS40, stream);
  run_conv<40, 20, 2, 1, false, false>(Bf[2], BS40, AP(3), BI(3), Bf[1], BS40, nullptr, 0, stream);
  run_conv<20, 40, 2, 1, true, false>(Bf[1], BS40, AP(4), BI(4), Bf[0], BS40, Bf[2], BS40, stream);
  unsigned short* E = Bf[0]; // enc1 output, 40 ch

  // encoder 2 on each half (shared weights)
  for (int h = 0; h < 2; ++h) {
    const unsigned short* he = E + (h ? 20L * TT : 0);
    unsigned short* CO = h ? CN_ : CS_;
    run_conv<20, 10, 1, 1, false, false>(he, BS40, AP(5), BI(5), Bf[1], BS40, nullptr, 0, stream);
    run_conv<10, 20, 1, 1, true, false>(Bf[1], BS40, AP(6), BI(6), Bf[2], BS40, he, BS40, stream);
    run_conv<20, 10, 2, 1, false, false>(Bf[2], BS40, AP(7), BI(7), Bf[1], BS40, nullptr, 0, stream);
    run_conv<10, 20, 2, 1, true, false>(Bf[1], BS40, AP(8), BI(8), Bf[3], BS40, Bf[2], BS40, stream);
    run_conv<20, 15, 1, 2, false, false>(Bf[3], BS40, AP(9), BI(9), CO, BS15, nullptr, 0, stream);
  }

  // soft VQ (prob -> d_out, quantized codes in place)
  vq_soft<<<dim3(TB / 256), dim3(256), 0, stream>>>(CS_, BS15, mean_s, prob_s);
  vq_soft<<<dim3(TB / 256), dim3(256), 0, stream>>>(CN_, BS15, mean_n, prob_n);

  // decoder on each quantized code (shared weights)
  for (int h = 0; h < 2; ++h) {
    unsigned short* Q = h ? CN_ : CS_;
    float* OH = h ? n_hat : s_hat;
    run_conv<15, 40, 1, 1, false, false>(Q, BS15, AP(10), BI(10), Bf[0], BS40, nullptr, 0, stream);
    run_conv<40, 20, 1, 1, false, false>(Bf[0], BS40, AP(11), BI(11), Bf[1], BS40, nullptr, 0, stream);
    run_conv<20, 40, 1, 1, true, false>(Bf[1], BS40, AP(12), BI(12), Bf[2], BS40, Bf[0], BS40, stream);
    run_conv<40, 20, 2, 1, false, false>(Bf[2], BS40, AP(13), BI(13), Bf[0], BS40, nullptr, 0, stream);
    run_conv<20, 40, 2, 1, true, false>(Bf[0], BS40, AP(14), BI(14), Bf[1], BS40, Bf[2], BS40, stream);
    run_conv<40, 20, 1, 1, false, false>(Bf[1], BS40, AP(15), BI(15), Bf[0], BS40, nullptr, 0, stream);
    run_conv<20, 40, 1, 1, true, false>(Bf[0], BS40, AP(16), BI(16), Bf[2], BS40, Bf[1], BS40, stream);
    run_conv<40, 20, 2, 1, false, false>(Bf[2], BS40, AP(17), BI(17), Bf[0], BS40, nullptr, 0, stream);
    run_conv<20, 40, 2, 1, true, false>(Bf[0], BS40, AP(18), BI(18), Bf[1], BS40, Bf[2], BS40, stream);
    run_conv<40, 1, 1, 0, false, true>(Bf[1], BS40, AP(19), BI(19), OH, BS1, nullptr, 0, stream);
  }
}
